// LstmWithCRFModel_60000693125757
// MI455X (gfx1250) — matrix operand order repaired and checked by reading, not compiled, not run on hardware
//
#include <hip/hip_runtime.h>
#include <hip/hip_bf16.h>

// ---------------- problem constants ----------------
#define Bq    32
#define Wq    128
#define BALLq 8
#define Eq    256
#define Hq    512
#define Cq    50
#define Vq    10000
constexpr int Nseq = Bq * BALLq;     // 256 sequences
constexpr int NW   = Nseq * Wq;      // 32768 (time-flattened rows)
constexpr int FH   = 4 * Hq;         // 2048 gate width
constexpr int H2   = 2 * Hq;         // 1024 concat width

typedef __attribute__((ext_vector_type(16))) __bf16 v16bf;
typedef __attribute__((ext_vector_type(8)))  float  v8f;
typedef __attribute__((ext_vector_type(4)))  int    v4i;

#ifndef __has_builtin
#define __has_builtin(x) 0
#endif

#if __has_builtin(__builtin_amdgcn_global_load_async_to_lds_b128)
#define HAVE_ASYNC_LDS 1
#else
#define HAVE_ASYNC_LDS 0
#endif

__device__ __forceinline__ float sigf(float x) { return 1.0f / (1.0f + __expf(-x)); }

// async 16B global -> LDS copy (CDNA5 GLOBAL_LOAD_ASYNC_TO_LDS_B128, ASYNCcnt)
__device__ __forceinline__ void cp_async16(void* lds, const void* glb)
{
#if HAVE_ASYNC_LDS
    __builtin_amdgcn_global_load_async_to_lds_b128(
        (__attribute__((address_space(1))) v4i*)glb,
        (__attribute__((address_space(3))) v4i*)lds,
        0, 0);
#else
    *(uint4*)lds = *(const uint4*)glb;
#endif
}

__device__ __forceinline__ void wait_async0()
{
#if __has_builtin(__builtin_amdgcn_s_wait_asynccnt)
    __builtin_amdgcn_s_wait_asynccnt(0);
#else
    asm volatile("s_wait_asynccnt 0x0" ::: "memory");
#endif
}

__device__ __forceinline__ void sched_fence()
{
#if __has_builtin(__builtin_amdgcn_sched_barrier)
    __builtin_amdgcn_sched_barrier(0);   // keep frag loads above, WMMAs below
#endif
}

// ---------------------------------------------------------------------------
// GEMM:  C[M x Nout] = A[M x K](bf16) * B[Nout x K](bf16)^T + bias
// Block: 128 threads (4 waves), 64x128 output tile, K stepped by 32.
// Double-buffered LDS filled by async global->LDS; 8 WMMAs/wave/K-step.
// M multiple of 64, K multiple of 32; Nout may be ragged (zero-padded/guarded).
// ---------------------------------------------------------------------------
__global__ __launch_bounds__(128) void gemm_bf16_wmma(
    const __bf16* __restrict__ A, const __bf16* __restrict__ Bm,
    float* __restrict__ Cm, const float* __restrict__ bias,
    int M, int Nout, int K, int ldc)
{
    __shared__ __bf16 sA[2][64][40];    // 80B rows: 16B-aligned chunks, bank spread
    __shared__ __bf16 sB[2][128][40];

    const int tid  = threadIdx.x;
    const int lane = tid & 31;
    const int wv   = tid >> 5;          // wave 0..3 -> A rows 16*wv..16*wv+15
    const int lo   = lane & 15;
    const int hi   = lane >> 4;

    const int bm = blockIdx.y * 64;
    const int bn = blockIdx.x * 128;

    const int arow    = tid >> 1;       // A staging row 0..63
    const int halfsel = tid & 1;        // which 16-elem half of the 32-wide K strip
    const int brow    = tid;            // B staging row 0..127
    const bool bvalid = (bn + brow) < Nout;

    // zero-fill padded B rows once (both buffers); never async-written later
    if (!bvalid) {
        const uint4 z = make_uint4(0u, 0u, 0u, 0u);
#pragma unroll
        for (int h = 0; h < 32; h += 8) {
            *(uint4*)&sB[0][brow][h] = z;
            *(uint4*)&sB[1][brow][h] = z;
        }
    }

    v8f acc[8];
#pragma unroll
    for (int j = 0; j < 8; ++j)
#pragma unroll
        for (int e = 0; e < 8; ++e) acc[j][e] = 0.0f;

    auto stage = [&](int buf, int k0) {
        const __bf16* ga = A + (size_t)(bm + arow) * K + k0 + halfsel * 16;
        cp_async16(&sA[buf][arow][halfsel * 16],     ga);
        cp_async16(&sA[buf][arow][halfsel * 16 + 8], ga + 8);
        if (bvalid) {
            const __bf16* gb = Bm + (size_t)(bn + brow) * K + k0;
            cp_async16(&sB[buf][brow][0],  gb);
            cp_async16(&sB[buf][brow][8],  gb + 8);
            cp_async16(&sB[buf][brow][16], gb + 16);
            cp_async16(&sB[buf][brow][24], gb + 24);
        }
    };

    int cur = 0;
    stage(0, 0);

    const int am = 16 * wv + lo;        // A fragment row for this wave
    const int kk = hi * 8;              // A frag K chunk base (ISA 16-bit A layout)
    const int ko = hi * 8;              // B frag K chunk base (same K order as A)

    for (int k0 = 0; k0 < K; k0 += 32) {
        wait_async0();                  // my fills of buffer `cur` landed
        __syncthreads();                // everyone's fills landed; prev reads done

        if (k0 + 32 < K) stage(cur ^ 1, k0 + 32);   // overlap next tile fill

        union U { uint4 u[2]; v16bf v; };
        U af;
        af.u[0] = *(const uint4*)&sA[cur][am][kk];
        af.u[1] = *(const uint4*)&sA[cur][am][kk + 16];

        U bf8[8];
#pragma unroll
        for (int j = 0; j < 8; ++j) {
            const int bnn = 16 * j + lo;
            bf8[j].u[0] = *(const uint4*)&sB[cur][bnn][ko];
            bf8[j].u[1] = *(const uint4*)&sB[cur][bnn][ko + 16];
        }

        sched_fence();                  // all 18 ds_loads issue before any WMMA

#pragma unroll
        for (int j = 0; j < 8; ++j)
            acc[j] = __builtin_amdgcn_wmma_f32_16x16x32_bf16(
                false, af.v, false, bf8[j].v, (short)0, acc[j], false, false);

        cur ^= 1;
    }

    // ---- store C (16x16 f32 layout: VGPR r -> M = 8*hi + r, N = lo) ----
    const int mbase = bm + 16 * wv + hi * 8;
#pragma unroll
    for (int j = 0; j < 8; ++j) {
        const int col = bn + 16 * j + lo;
        if (col < Nout) {
            const float bv = bias ? bias[col] : 0.0f;
#pragma unroll
            for (int rr = 0; rr < 8; ++rr)
                Cm[(size_t)(mbase + rr) * ldc + col] = acc[j][rr] + bv;
        }
    }
}

// ---------------------------------------------------------------------------
// Embedding gather + fp32 -> bf16.  x[n,w,e], n = ball*B + b (moveaxis(2,0)).
// ---------------------------------------------------------------------------
__global__ void embed_kernel(const int* __restrict__ inputs,
                             const float* __restrict__ emb,
                             __bf16* __restrict__ xout)
{
    const size_t id = (size_t)blockIdx.x * blockDim.x + threadIdx.x;
    if (id >= (size_t)Nseq * Wq * Eq) return;
    const int e = (int)(id % Eq);
    const int w = (int)((id / Eq) % Wq);
    const int n = (int)(id / ((size_t)Eq * Wq));
    const int ball = n / Bq;
    const int b    = n % Bq;
    const int tok  = inputs[((size_t)b * Wq + w) * BALLq + ball];
    xout[id] = (__bf16)emb[(size_t)tok * Eq + e];
}

__global__ void cvt_kernel(const float* __restrict__ s, __bf16* __restrict__ d, int cnt)
{
    const int i = blockIdx.x * blockDim.x + threadIdx.x;
    if (i < cnt) d[i] = (__bf16)s[i];
}

__global__ void zero_kernel(unsigned* __restrict__ p, int cnt)
{
    const int i = blockIdx.x * blockDim.x + threadIdx.x;
    if (i < cnt) p[i] = 0u;
}

// ---------------------------------------------------------------------------
// LSTM cell (4-wide vectorized): g = xs[:,t,:] + h_prev@Whh^T; gates -> c,h.
// ---------------------------------------------------------------------------
__global__ void lstm_cell_kernel(const float* __restrict__ xs,
                                 const float* __restrict__ grec,
                                 float* __restrict__ c,
                                 __bf16* __restrict__ hstate,
                                 __bf16* __restrict__ hout,
                                 int t, int outStride, int outOfs)
{
    const int id = blockIdx.x * blockDim.x + threadIdx.x;
    if (id >= Nseq * Hq / 4) return;
    const int n = id / (Hq / 4);
    const int j = (id % (Hq / 4)) * 4;
    const size_t xb = ((size_t)n * Wq + t) * FH;
    const size_t gb = (size_t)n * FH;

    const float4 xi = *(const float4*)&xs[xb + j];
    const float4 xf = *(const float4*)&xs[xb + Hq + j];
    const float4 xg = *(const float4*)&xs[xb + 2 * Hq + j];
    const float4 xo = *(const float4*)&xs[xb + 3 * Hq + j];
    const float4 ri = *(const float4*)&grec[gb + j];
    const float4 rf = *(const float4*)&grec[gb + Hq + j];
    const float4 rg = *(const float4*)&grec[gb + 2 * Hq + j];
    const float4 ro = *(const float4*)&grec[gb + 3 * Hq + j];
    float4 cv = *(const float4*)&c[(size_t)n * Hq + j];

    union { __bf16 h[4]; uint2 u; } hb;
    float hv[4];
    {
        const float gi[4] = { xi.x + ri.x, xi.y + ri.y, xi.z + ri.z, xi.w + ri.w };
        const float gf[4] = { xf.x + rf.x, xf.y + rf.y, xf.z + rf.z, xf.w + rf.w };
        const float gg[4] = { xg.x + rg.x, xg.y + rg.y, xg.z + rg.z, xg.w + rg.w };
        const float go[4] = { xo.x + ro.x, xo.y + ro.y, xo.z + ro.z, xo.w + ro.w };
        float cc[4] = { cv.x, cv.y, cv.z, cv.w };
#pragma unroll
        for (int q = 0; q < 4; ++q) {
            cc[q] = sigf(gf[q]) * cc[q] + sigf(gi[q]) * tanhf(gg[q]);
            hv[q] = sigf(go[q]) * tanhf(cc[q]);
            hb.h[q] = (__bf16)hv[q];
        }
        cv.x = cc[0]; cv.y = cc[1]; cv.z = cc[2]; cv.w = cc[3];
    }
    *(float4*)&c[(size_t)n * Hq + j] = cv;
    *(uint2*)&hstate[(size_t)n * Hq + j] = hb.u;
    *(uint2*)&hout[(size_t)n * Wq * outStride + (size_t)t * outStride + outOfs + j] = hb.u;
}

// ---------------------------------------------------------------------------
// CRF forward (log-partition) + gold sequence score -> loss += score - logZ
// ---------------------------------------------------------------------------
__global__ __launch_bounds__(64) void crf_forward_kernel(
    const float* __restrict__ outputs, const float* __restrict__ trans,
    const int* __restrict__ tags, const int* __restrict__ seqlen,
    float* __restrict__ loss)
{
    __shared__ float tr[Cq * Cq];
    __shared__ float alpha[Cq], nalpha[Cq];
    const int n = blockIdx.x, tid = threadIdx.x;
    for (int i = tid; i < Cq * Cq; i += 64) tr[i] = trans[i];
    const float* em = outputs + (size_t)n * Wq * Cq;
    const int L = seqlen[n];
    if (tid < Cq) alpha[tid] = em[tid];
    __syncthreads();
    for (int t = 1; t < Wq; ++t) {
        if (tid < Cq) {
            float m = -3.4e38f;
            for (int i = 0; i < Cq; ++i) m = fmaxf(m, alpha[i] + tr[i * Cq + tid]);
            float s = 0.0f;
            for (int i = 0; i < Cq; ++i) s += __expf(alpha[i] + tr[i * Cq + tid] - m);
            const float nv = m + __logf(s) + em[t * Cq + tid];
            nalpha[tid] = (t < L) ? nv : alpha[tid];
        }
        __syncthreads();
        if (tid < Cq) alpha[tid] = nalpha[tid];
        __syncthreads();
    }
    if (tid == 0) {
        float m = -3.4e38f;
        for (int i = 0; i < Cq; ++i) m = fmaxf(m, alpha[i]);
        float s = 0.0f;
        for (int i = 0; i < Cq; ++i) s += __expf(alpha[i] - m);
        const float logZ = m + __logf(s);
        const int* tg = tags + (size_t)n * Wq;
        float score = em[tg[0]];
        for (int t = 1; t < Wq; ++t)
            if (t < L) score += em[t * Cq + tg[t]] + tr[tg[t - 1] * Cq + tg[t]];
        atomicAdd(loss, score - logZ);
    }
}

// ---------------------------------------------------------------------------
// Viterbi (reference variant: logsumexp update, argmax backpointers).
// ---------------------------------------------------------------------------
__global__ __launch_bounds__(64) void viterbi_kernel(
    const float* __restrict__ outputs, const float* __restrict__ trans,
    const int* __restrict__ seqlen, int* __restrict__ bps,
    float* __restrict__ pred, float* __restrict__ score)
{
    __shared__ float tr[Cq * Cq];
    __shared__ float vit[Cq], nvit[Cq];
    const int n = blockIdx.x, tid = threadIdx.x;
    for (int i = tid; i < Cq * Cq; i += 64) tr[i] = trans[i];
    const float* em = outputs + (size_t)n * Wq * Cq;
    const int L = seqlen[n];
    int* bp = bps + (size_t)n * (Wq - 1) * Cq;
    if (tid < Cq) vit[tid] = em[tid];
    __syncthreads();
    for (int t = 1; t < Wq; ++t) {
        if (tid < Cq) {
            float m = -3.4e38f; int am = 0;
            for (int i = 0; i < Cq; ++i) {
                const float v = vit[i] + tr[i * Cq + tid];
                if (v > m) { m = v; am = i; }
            }
            float s = 0.0f;
            for (int i = 0; i < Cq; ++i) s += __expf(vit[i] + tr[i * Cq + tid] - m);
            const float nv = m + __logf(s) + em[t * Cq + tid];
            const bool valid = (t < L);
            nvit[tid] = valid ? nv : vit[tid];
            bp[(t - 1) * Cq + tid] = valid ? am : tid;
        }
        __syncthreads();
        if (tid < Cq) vit[tid] = nvit[tid];
        __syncthreads();
    }
    if (tid == 0) {
        int best = 0; float bm = vit[0];
        for (int i = 1; i < Cq; ++i) if (vit[i] > bm) { bm = vit[i]; best = i; }
        score[n] = bm;
        int cur = best;
        pred[(size_t)n * Wq + (Wq - 1)] = (float)best;
        for (int t = Wq - 2; t >= 0; --t) {
            cur = bp[t * Cq + cur];
            pred[(size_t)n * Wq + t] = (float)cur;
        }
    }
}

// ---------------------------------------------------------------------------
extern "C" void kernel_launch(void* const* d_in, const int* in_sizes, int n_in,
                              void* d_out, int out_size, void* d_ws, size_t ws_size,
                              hipStream_t stream)
{
    const int*   inputs = (const int*)d_in[0];
    const int*   tags   = (const int*)d_in[1];
    const int*   seqlen = (const int*)d_in[2];
    const float* emb    = (const float*)d_in[3];
    const float* Wih1   = (const float*)d_in[4];
    const float* Whh1   = (const float*)d_in[5];
    const float* b1     = (const float*)d_in[6];
    const float* Wih2f  = (const float*)d_in[7];
    const float* Whh2f  = (const float*)d_in[8];
    const float* b2f    = (const float*)d_in[9];
    const float* Wih2b  = (const float*)d_in[10];
    const float* Whh2b  = (const float*)d_in[11];
    const float* b2b    = (const float*)d_in[12];
    const float* Wd     = (const float*)d_in[13];
    const float* bd     = (const float*)d_in[14];
    const float* trans  = (const float*)d_in[15];

    float* out = (float*)d_out;
    const size_t OUT_LOSS  = (size_t)NW * Cq;          // 1,638,400
    const size_t OUT_PRED  = OUT_LOSS + 1;
    const size_t OUT_SCORE = OUT_PRED + (size_t)NW;

    // ---- workspace carve-up ----
    char* ws = (char*)d_ws;
    __bf16* xBf   = (__bf16*)ws;                 ws += (size_t)NW * Eq * 2;
    float*  xs    = (float*)ws;                  ws += (size_t)NW * FH * 4;   // reused x3
    __bf16* h1    = (__bf16*)ws;                 ws += (size_t)NW * Hq * 2;
    __bf16* h2    = (__bf16*)ws;                 ws += (size_t)NW * H2 * 2;
    __bf16* wWih1 = (__bf16*)ws;                 ws += (size_t)FH * Eq * 2;
    __bf16* wWhh1 = (__bf16*)ws;                 ws += (size_t)FH * Hq * 2;
    __bf16* wWih2f= (__bf16*)ws;                 ws += (size_t)FH * Hq * 2;
    __bf16* wWhh2f= (__bf16*)ws;                 ws += (size_t)FH * Hq * 2;
    __bf16* wWih2b= (__bf16*)ws;                 ws += (size_t)FH * Hq * 2;
    __bf16* wWhh2b= (__bf16*)ws;                 ws += (size_t)FH * Hq * 2;
    __bf16* wWd   = (__bf16*)ws;                 ws += (size_t)Cq * H2 * 2;
    __bf16* hstate= (__bf16*)ws;                 ws += (size_t)Nseq * Hq * 2;
    float*  cstate= (float*)ws;                  ws += (size_t)Nseq * Hq * 4;
    float*  gbuf  = (float*)ws;                  ws += (size_t)Nseq * FH * 4;
    int*    bps   = (int*)ws;                    ws += (size_t)Nseq * (Wq - 1) * Cq * 4;

    auto cvt = [&](const float* s, __bf16* d, int cnt) {
        cvt_kernel<<<(cnt + 255) / 256, 256, 0, stream>>>(s, d, cnt);
    };
    auto zero = [&](void* p, size_t bytes) {
        int cnt = (int)(bytes / 4);
        zero_kernel<<<(cnt + 255) / 256, 256, 0, stream>>>((unsigned*)p, cnt);
    };
    auto gemm = [&](const __bf16* A, const __bf16* Bm, float* Cm, const float* bias,
                    int M, int Nout, int K, int ldc) {
        dim3 g((Nout + 127) / 128, M / 64);
        gemm_bf16_wmma<<<g, 128, 0, stream>>>(A, Bm, Cm, bias, M, Nout, K, ldc);
    };
    auto cell = [&](__bf16* hout, int t, int outStride, int outOfs) {
        lstm_cell_kernel<<<(Nseq * Hq / 4 + 255) / 256, 256, 0, stream>>>(
            xs, gbuf, cstate, hstate, hout, t, outStride, outOfs);
    };

    // weight conversions (fp32 -> bf16)
    cvt(Wih1,  wWih1,  FH * Eq);
    cvt(Whh1,  wWhh1,  FH * Hq);
    cvt(Wih2f, wWih2f, FH * Hq);
    cvt(Whh2f, wWhh2f, FH * Hq);
    cvt(Wih2b, wWih2b, FH * Hq);
    cvt(Whh2b, wWhh2b, FH * Hq);
    cvt(Wd,    wWd,    Cq * H2);

    // embedding gather
    {
        const size_t tot = (size_t)NW * Eq;
        embed_kernel<<<(unsigned)((tot + 255) / 256), 256, 0, stream>>>(inputs, emb, xBf);
    }

    // ---------- layer 1 ----------
    gemm(xBf, wWih1, xs, b1, NW, FH, Eq, FH);
    zero(hstate, (size_t)Nseq * Hq * 2);
    zero(cstate, (size_t)Nseq * Hq * 4);
    for (int t = 0; t < Wq; ++t) {
        gemm(hstate, wWhh1, gbuf, nullptr, Nseq, FH, Hq, FH);
        cell(h1, t, Hq, 0);
    }

    // ---------- layer 2 forward ----------
    gemm(h1, wWih2f, xs, b2f, NW, FH, Hq, FH);
    zero(hstate, (size_t)Nseq * Hq * 2);
    zero(cstate, (size_t)Nseq * Hq * 4);
    for (int t = 0; t < Wq; ++t) {
        gemm(hstate, wWhh2f, gbuf, nullptr, Nseq, FH, Hq, FH);
        cell(h2, t, H2, 0);
    }

    // ---------- layer 2 backward ----------
    gemm(h1, wWih2b, xs, b2b, NW, FH, Hq, FH);
    zero(hstate, (size_t)Nseq * Hq * 2);
    zero(cstate, (size_t)Nseq * Hq * 4);
    for (int t = Wq - 1; t >= 0; --t) {
        gemm(hstate, wWhh2b, gbuf, nullptr, Nseq, FH, Hq, FH);
        cell(h2, t, H2, Hq);
    }

    // ---------- dense head: outputs -> d_out[0 .. NW*C) ----------
    gemm(h2, wWd, out, bd, NW, Cq, H2, Cq);

    // ---------- CRF loss + viterbi ----------
    zero(out + OUT_LOSS, 4);
    crf_forward_kernel<<<Nseq, 64, 0, stream>>>(out, trans, tags, seqlen, out + OUT_LOSS);
    viterbi_kernel<<<Nseq, 64, 0, stream>>>(out, trans, seqlen, bps,
                                            out + OUT_PRED, out + OUT_SCORE);
    (void)in_sizes; (void)n_in; (void)out_size; (void)ws_size;
}
